// CausalSelfAttention_75831942578784
// MI455X (gfx1250) — matrix operand order repaired and checked by reading, not compiled, not run on hardware
//
#include <hip/hip_runtime.h>

typedef __attribute__((ext_vector_type(16))) __bf16 v16bf;
typedef __attribute__((ext_vector_type(8)))  __bf16 v8bf;
typedef __attribute__((ext_vector_type(8)))  float  v8f;

#define S_ 2048
#define D_ 1024
#define H_ 16

__device__ __forceinline__ v16bf cat8(v8bf lo, v8bf hi) {
  v16bf r;
#pragma unroll
  for (int i = 0; i < 8; ++i) { r[i] = lo[i]; r[i + 8] = hi[i]; }
  return r;
}

__global__ void cvt_bf16_kernel(const float* __restrict__ src,
                                __bf16* __restrict__ dst, int n) {
  int i = blockIdx.x * blockDim.x + threadIdx.x;
  if (i < n) dst[i] = (__bf16)src[i];
}

// One fused GEMM kernel, C = A[S,D] @ W[N,D]^T in 16x64 per-wave tiles.
// mode 0: q  -> rmsnorm + rope -> bf16 [S,D]
// mode 1: k  -> rmsnorm + rope -> bf16 [S,D]
// mode 2: v  -> lambda-mix with v1 -> bf16 transposed [D,S]
// mode 3: out GEMM -> fp32 [S,D] to d_out
__global__ __launch_bounds__(128) void gemm_fused_kernel(
    const __bf16* __restrict__ A,
    const __bf16* __restrict__ W0, const __bf16* __restrict__ W1,
    const __bf16* __restrict__ W2,
    int mode_base,
    const float* __restrict__ v1, const float* __restrict__ lam,
    __bf16* __restrict__ qb, __bf16* __restrict__ kbuf,
    __bf16* __restrict__ vT, float* __restrict__ outp) {
  const int mode = mode_base + blockIdx.z;
  const __bf16* __restrict__ W = (mode == 1) ? W1 : (mode == 2) ? W2 : W0;
  const int lane = threadIdx.x & 31;
  const int wv = threadIdx.x >> 5;
  const int hh = lane >> 4;     // lane half selects K sub-range per ISA layouts
  const int nl = lane & 15;     // M for A, N for B/C
  const int row0 = (blockIdx.x * 4 + wv) * 16;
  const int col0 = blockIdx.y * 64;

  v8f acc[4];
#pragma unroll
  for (int t = 0; t < 4; ++t)
#pragma unroll
    for (int j = 0; j < 8; ++j) acc[t][j] = 0.f;

  for (int kk = 0; kk < D_; kk += 32) {
    // A tile 16x32 bf16: K = 16*hh+0..15 (contiguous, matches B)
    const __bf16* ab = A + (size_t)(row0 + nl) * D_ + kk + 16 * hh;
    v16bf av = cat8(*(const v8bf*)ab, *(const v8bf*)(ab + 8));
#pragma unroll
    for (int t = 0; t < 4; ++t) {
      // B tile 32x16 bf16: lane = N (column of W^T = row of W), K = kk+16*hh+0..15
      const __bf16* bb = W + (size_t)(col0 + 16 * t + nl) * D_ + kk + 16 * hh;
      v16bf bv = *(const v16bf*)bb;
      acc[t] = __builtin_amdgcn_wmma_f32_16x16x32_bf16(
          false, av, false, bv, (short)0, acc[t], false, false);
    }
  }

  if (mode < 2) {
    __bf16* __restrict__ dst = (mode == 0) ? qb : kbuf;
    const float kRopeC = -0.28782313662425572f;  // -ln(10000)/32
#pragma unroll
    for (int j = 0; j < 8; ++j) {
      // RMS norm over the 64-col head: N lives across the 16-lane half.
      float ss = 0.f;
#pragma unroll
      for (int t = 0; t < 4; ++t) ss += acc[t][j] * acc[t][j];
      ss += __shfl_xor(ss, 1); ss += __shfl_xor(ss, 2);
      ss += __shfl_xor(ss, 4); ss += __shfl_xor(ss, 8);
      float r = rsqrtf(ss * (1.0f / 64.0f) + 1.1920929e-07f);
#pragma unroll
      for (int t = 0; t < 4; ++t) acc[t][j] *= r;
      // RoPE: col n pairs with n+32 => acc[t] with acc[t+2], same lane.
      const int srow = row0 + j + 8 * hh;
#pragma unroll
      for (int t = 0; t < 2; ++t) {
        float fi = (float)(nl + 16 * t);
        float ang = (float)srow * __expf(kRopeC * fi);
        float c = __cosf(ang), sn = __sinf(ang);
        float a = acc[t][j], b = acc[t + 2][j];
        acc[t][j] = a * c + b * sn;
        acc[t + 2][j] = b * c - a * sn;
      }
#pragma unroll
      for (int t = 0; t < 4; ++t)
        dst[(size_t)srow * D_ + col0 + 16 * t + nl] = (__bf16)acc[t][j];
    }
  } else if (mode == 2) {
    const float l = *lam;
#pragma unroll
    for (int j = 0; j < 8; ++j) {
      const int srow = row0 + j + 8 * hh;
#pragma unroll
      for (int t = 0; t < 4; ++t) {
        const int c = col0 + 16 * t + nl;
        float vv = (1.0f - l) * acc[t][j] + l * v1[(size_t)srow * D_ + c];
        vT[(size_t)c * S_ + srow] = (__bf16)vv;  // transposed for PV GEMM
      }
    }
  } else {
#pragma unroll
    for (int j = 0; j < 8; ++j) {
      const int srow = row0 + j + 8 * hh;
#pragma unroll
      for (int t = 0; t < 4; ++t)
        outp[(size_t)srow * D_ + col0 + 16 * t + nl] = acc[t][j];
    }
  }
}

// Flash attention: each wave owns 16 queries of one head, online softmax
// over 32-key blocks. S = Q(16x64) K^T(64x32) via 4 WMMA; P transposed
// C-layout -> A-layout through per-wave LDS; O += P(16x32) V(32x64) via 4 WMMA.
__global__ __launch_bounds__(128) void attn_kernel(
    const __bf16* __restrict__ qb, const __bf16* __restrict__ kbuf,
    const __bf16* __restrict__ vT, __bf16* __restrict__ yb) {
  __shared__ __align__(32) __bf16 lds[4][16 * 32];
  const int lane = threadIdx.x & 31;
  const int wv = threadIdx.x >> 5;
  const int hh = lane >> 4;
  const int nl = lane & 15;
  const int h = blockIdx.y;
  const int q0 = (blockIdx.x * 4 + wv) * 16;

  v16bf qa[2];
#pragma unroll
  for (int s = 0; s < 2; ++s) {
    const __bf16* base = qb + (size_t)(q0 + nl) * D_ + h * 64 + 32 * s + 16 * hh;
    qa[s] = cat8(*(const v8bf*)base, *(const v8bf*)(base + 8));
  }

  v8f O[4];
  float mrun[8], lrun[8];
#pragma unroll
  for (int j = 0; j < 8; ++j) {
    mrun[j] = -1e30f; lrun[j] = 0.f;
    O[0][j] = 0.f; O[1][j] = 0.f; O[2][j] = 0.f; O[3][j] = 0.f;
  }

  const int nkb = (q0 + 47) >> 5;  // key blocks covering keys 0..q0+15
  for (int kbk = 0; kbk < nkb; ++kbk) {
    const int kpos0 = kbk * 32;
    v8f sacc[2];
#pragma unroll
    for (int j = 0; j < 8; ++j) { sacc[0][j] = 0.f; sacc[1][j] = 0.f; }
#pragma unroll
    for (int s = 0; s < 2; ++s) {
#pragma unroll
      for (int t = 0; t < 2; ++t) {
        const __bf16* base =
            kbuf + (size_t)(kpos0 + 16 * t + nl) * D_ + h * 64 + 32 * s + 16 * hh;
        v16bf bv = *(const v16bf*)base;
        sacc[t] = __builtin_amdgcn_wmma_f32_16x16x32_bf16(
            false, qa[s], false, bv, (short)0, sacc[t], false, false);
      }
    }
    const bool need_mask = (kpos0 + 31) > q0;
#pragma unroll
    for (int t = 0; t < 2; ++t) {
#pragma unroll
      for (int j = 0; j < 8; ++j) {
        float v = sacc[t][j] * 0.125f;  // 1/sqrt(64)
        if (need_mask) {
          int key = kpos0 + 16 * t + nl;
          int qr = q0 + j + 8 * hh;
          if (key > qr) v = -1e30f;
        }
        sacc[t][j] = v;
      }
    }
#pragma unroll
    for (int j = 0; j < 8; ++j) {
      float mx = fmaxf(sacc[0][j], sacc[1][j]);
      mx = fmaxf(mx, __shfl_xor(mx, 1));
      mx = fmaxf(mx, __shfl_xor(mx, 2));
      mx = fmaxf(mx, __shfl_xor(mx, 4));
      mx = fmaxf(mx, __shfl_xor(mx, 8));
      float mnew = fmaxf(mrun[j], mx);
      float corr = __expf(mrun[j] - mnew);
      float p0 = __expf(sacc[0][j] - mnew);
      float p1 = __expf(sacc[1][j] - mnew);
      sacc[0][j] = p0; sacc[1][j] = p1;
      float rs = p0 + p1;
      rs += __shfl_xor(rs, 1); rs += __shfl_xor(rs, 2);
      rs += __shfl_xor(rs, 4); rs += __shfl_xor(rs, 8);
      lrun[j] = lrun[j] * corr + rs;
      mrun[j] = mnew;
      O[0][j] *= corr; O[1][j] *= corr; O[2][j] *= corr; O[3][j] *= corr;
    }
    // P: C-layout scatter into LDS, reload in A-layout (per-wave region).
#pragma unroll
    for (int j = 0; j < 8; ++j) {
      lds[wv][(j + 8 * hh) * 32 + nl] = (__bf16)sacc[0][j];
      lds[wv][(j + 8 * hh) * 32 + 16 + nl] = (__bf16)sacc[1][j];
    }
    asm volatile("s_wait_dscnt 0" ::: "memory");
    const __bf16* pl = &lds[wv][nl * 32 + 16 * hh];
    v16bf pa = cat8(*(const v8bf*)pl, *(const v8bf*)(pl + 8));
#pragma unroll
    for (int t2 = 0; t2 < 4; ++t2) {
      const __bf16* base =
          vT + (size_t)(h * 64 + 16 * t2 + nl) * S_ + kpos0 + 16 * hh;
      v16bf bv = *(const v16bf*)base;
      O[t2] = __builtin_amdgcn_wmma_f32_16x16x32_bf16(
          false, pa, false, bv, (short)0, O[t2], false, false);
    }
  }
#pragma unroll
  for (int j = 0; j < 8; ++j) {
    float inv = 1.0f / lrun[j];
    const int srow = q0 + j + 8 * hh;
#pragma unroll
    for (int t = 0; t < 4; ++t)
      yb[(size_t)srow * D_ + h * 64 + 16 * t + nl] = (__bf16)(O[t][j] * inv);
  }
}

extern "C" void kernel_launch(void* const* d_in, const int* in_sizes, int n_in,
                              void* d_out, int out_size, void* d_ws, size_t ws_size,
                              hipStream_t stream) {
  const float* x    = (const float*)d_in[0];
  const float* v1   = (const float*)d_in[1];
  const float* Wq   = (const float*)d_in[2];
  const float* Wk   = (const float*)d_in[3];
  const float* Wv   = (const float*)d_in[4];
  const float* Wout = (const float*)d_in[5];
  const float* lam  = (const float*)d_in[6];
  float* out = (float*)d_out;

  // Workspace layout (bf16), total 28 MB:
  __bf16* xb  = (__bf16*)d_ws;
  __bf16* wqb = xb  + (size_t)S_ * D_;
  __bf16* wkb = wqb + (size_t)D_ * D_;
  __bf16* wvb = wkb + (size_t)D_ * D_;
  __bf16* wob = wvb + (size_t)D_ * D_;
  __bf16* qb  = wob + (size_t)D_ * D_;
  __bf16* kbf = qb  + (size_t)S_ * D_;
  __bf16* vT  = kbf + (size_t)S_ * D_;
  __bf16* yb  = vT  + (size_t)S_ * D_;

  const int nx = S_ * D_, nw = D_ * D_;
  cvt_bf16_kernel<<<(nx + 255) / 256, 256, 0, stream>>>(x, xb, nx);
  cvt_bf16_kernel<<<(nw + 255) / 256, 256, 0, stream>>>(Wq, wqb, nw);
  cvt_bf16_kernel<<<(nw + 255) / 256, 256, 0, stream>>>(Wk, wkb, nw);
  cvt_bf16_kernel<<<(nw + 255) / 256, 256, 0, stream>>>(Wv, wvb, nw);
  cvt_bf16_kernel<<<(nw + 255) / 256, 256, 0, stream>>>(Wout, wob, nw);

  // QKV projections + fused rmsnorm/rope/v-mix epilogues.
  gemm_fused_kernel<<<dim3(S_ / 64, D_ / 64, 3), 128, 0, stream>>>(
      xb, wqb, wkb, wvb, 0, v1, lam, qb, kbf, vT, out);

  // Causal flash attention.
  attn_kernel<<<dim3(S_ / 64, H_), 128, 0, stream>>>(qb, kbf, vT, yb);

  // Output projection -> fp32 d_out (first output of the tuple).
  gemm_fused_kernel<<<dim3(S_ / 64, D_ / 64, 1), 128, 0, stream>>>(
      yb, wob, wob, wob, 3, v1, lam, qb, kbf, vT, out);

  // Second tuple output: v1 passthrough.
  hipMemcpyAsync(out + (size_t)S_ * D_, v1, (size_t)S_ * D_ * sizeof(float),
                 hipMemcpyDeviceToDevice, stream);
}
